// MultiHeadAttention_81664508166582
// MI455X (gfx1250) — matrix operand order repaired and checked by reading, not compiled, not run on hardware
//
#include <hip/hip_runtime.h>

// ---------------------------------------------------------------------------
// Types for CDNA5 WMMA
// ---------------------------------------------------------------------------
typedef __bf16 bf16_t;
typedef __attribute__((ext_vector_type(16))) __bf16 v16bf;
typedef __attribute__((ext_vector_type(8)))  float  v8f;

#define BS 8
#define T  1024
#define E  512
#define H  8
#define HD 64
#define CU 6

// LDS partition (bytes) for the fused attention kernel
#define SM_U    0                      // bf16 u strip [CU][16][1024] = 196608
#define SM_S    196608                 // f32 score tile [16][1024]   =  65536
#define SM_PA   262144                 // f32 PV partials [8][16][16] =   8192
#define SM_MSK  270336                 // u8 mask strip [16][1024]    =  16384
#define SM_TOT  286720

// ---------------------------------------------------------------------------
// Kernel 0a: f32 -> bf16 elementwise (for x)
// ---------------------------------------------------------------------------
__global__ void cvt_x_kernel(const float* __restrict__ x, bf16_t* __restrict__ xb) {
    size_t i = (size_t)blockIdx.x * blockDim.x + threadIdx.x;   // exactly N threads
    xb[i] = (bf16_t)x[i];
}

// ---------------------------------------------------------------------------
// Kernel 0b: f32 [f][e] -> bf16 transposed [e][f]  (512x512 weights)
// ---------------------------------------------------------------------------
__global__ void cvt_wT_kernel(const float* __restrict__ w, bf16_t* __restrict__ wT) {
    int f = blockIdx.x * 16 + (threadIdx.x & 15);
    int e = blockIdx.y * 16 + (threadIdx.x >> 4);     // blockDim = 256 -> 16x16 tile
    wT[(size_t)e * E + f] = (bf16_t)w[(size_t)f * E + e];
}

// ---------------------------------------------------------------------------
// Kernel 1: projection GEMM  C[8192,512] = xb[8192,512] @ wT[512,512]
// mode 0: store [b][h][t][d] (Q, V)     mode 1: store [b][h][d][t] (K transposed)
// ---------------------------------------------------------------------------
__global__ void proj_kernel(const bf16_t* __restrict__ A, const bf16_t* __restrict__ BT,
                            bf16_t* __restrict__ out, int mode) {
    const int lane = threadIdx.x & 31;
    const int wave = threadIdx.x >> 5;
    const int m0 = blockIdx.x * 16;
    const int n0 = (blockIdx.y * 8 + wave) * 16;
    const int row = lane & 15;
    const int hi  = lane >> 4;

    v8f acc = {};
    for (int k0 = 0; k0 < E; k0 += 32) {
        v16bf a, b;
        const bf16_t* ap = A + (size_t)(m0 + row) * E + k0 + hi * 8;
        __builtin_prefetch(ap + 32, 0, 0);
#pragma unroll
        for (int j = 0; j < 8; ++j) { a[j] = ap[j]; a[j + 8] = ap[16 + j]; }
        const bf16_t* bp = BT + (size_t)(k0 + hi * 8) * E + n0 + row;
#pragma unroll
        for (int j = 0; j < 8; ++j) { b[j] = bp[(size_t)j * E]; b[j + 8] = bp[(size_t)(16 + j) * E]; }
        acc = __builtin_amdgcn_wmma_f32_16x16x32_bf16(false, a, false, b,
                                                      (short)0, acc, false, false);
    }

    const int col = n0 + (lane & 15);
    const int h = col >> 6, d = col & 63;
#pragma unroll
    for (int r = 0; r < 8; ++r) {
        int m = m0 + r + hi * 8;
        int bb = m >> 10, t = m & (T - 1);
        size_t idx = (mode == 0)
            ? ((((size_t)bb * H + h) * T + t) * HD + d)      // [b][h][t][d]
            : ((((size_t)bb * H + h) * HD + d) * T + t);     // [b][h][d][t]
        out[idx] = (bf16_t)acc[r];
    }
}

// ---------------------------------------------------------------------------
// Kernel 2 (mega-fused attention): per block = (b, q-tile of 16), loop heads.
// Everything stays in LDS: u strip (bf16, read from HBM exactly once), score
// tile (f32), mask strip, PV partials. Normalized attn is the only attention
// HBM write; raw scores never touch global memory.
// blockDim = 256 (8 waves). Dynamic LDS = 280 KB (<= 320 KB/WG on CDNA5).
// ---------------------------------------------------------------------------
__global__ void __launch_bounds__(256)
attn_fused_kernel(const bf16_t* __restrict__ Qb, const bf16_t* __restrict__ Ktb,
                  const bf16_t* __restrict__ Vb,
                  const float* __restrict__ u, const unsigned char* __restrict__ umask,
                  const float* __restrict__ wu, const float* __restrict__ bu,
                  float* __restrict__ attn, bf16_t* __restrict__ ctxb) {
    extern __shared__ char smem[];
    bf16_t*        u_s = (bf16_t*)(smem + SM_U);    // [CU][16][1024]
    float*         s   = (float*)(smem + SM_S);     // [16][1024]
    float*         pa  = (float*)(smem + SM_PA);    // [8][16][16]
    unsigned char* m_s = (unsigned char*)(smem + SM_MSK); // [16][1024]

    const int tid  = threadIdx.x;
    const int lane = tid & 31;
    const int wave = tid >> 5;
    const int q0 = blockIdx.x * 16;
    const int b  = blockIdx.y;
    const int row = lane & 15;
    const int hi  = lane >> 4;

    // --- one-time: u strip (f32 -> bf16) and mask strip into LDS -----------
    for (int i = tid; i < CU * 16 * 1024; i += 256) {
        int c = i >> 14, rem = i & 16383, r = rem >> 10, k = rem & 1023;
        u_s[i] = (bf16_t)u[(((size_t)b * CU + c) * T + (q0 + r)) * T + k];
    }
    for (int i = tid; i < 16 * 1024; i += 256) {
        int k = i >> 4, r = i & 15;          // 16-byte coalesced runs in global
        m_s[r * 1024 + k] = umask[((size_t)b * T + k) * T + (q0 + r)];
    }
    __syncthreads();

    for (int h = 0; h < H; ++h) {
        const float buh = bu[h];
        float wuh[CU];
#pragma unroll
        for (int c = 0; c < CU; ++c) wuh[c] = wu[h * CU + c];
        const bf16_t* Qp = Qb  + (((size_t)b * H + h) * T) * HD;
        const bf16_t* Kp = Ktb + (((size_t)b * H + h) * HD) * T;
        const bf16_t* Vp = Vb  + (((size_t)b * H + h) * T) * HD;

        // 1) ubias (+mask) pass into s
        for (int i = tid; i < 16 * 1024; i += 256) {
            int r = i >> 10, k = i & 1023;
            float bias = buh;
#pragma unroll
            for (int c = 0; c < CU; ++c)
                bias += (float)u_s[(c * 16 + r) * 1024 + k] * wuh[c];
            if (m_s[i]) bias = -__builtin_inff();   // m_s layout matches [r][k]
            s[i] = bias;
        }
        __syncthreads();

        // 2) scores: s += 8 * (Q K^T).  Q A-fragments loaded once per head.
        v16bf aq0, aq1;
        {
            const bf16_t* ap = Qp + (size_t)(q0 + row) * HD + hi * 8;
#pragma unroll
            for (int j = 0; j < 8; ++j) { aq0[j] = ap[j];      aq0[j + 8] = ap[16 + j];
                                          aq1[j] = ap[32 + j]; aq1[j + 8] = ap[48 + j]; }
        }
        for (int kt = wave * 8; kt < wave * 8 + 8; ++kt) {
            v16bf b0, b1;
            const bf16_t* bp0 = Kp + (size_t)(hi * 8) * T + kt * 16 + row;
            const bf16_t* bp1 = Kp + (size_t)(32 + hi * 8) * T + kt * 16 + row;
#pragma unroll
            for (int j = 0; j < 8; ++j) { b0[j] = bp0[(size_t)j * T]; b0[j + 8] = bp0[(size_t)(16 + j) * T];
                                          b1[j] = bp1[(size_t)j * T]; b1[j + 8] = bp1[(size_t)(16 + j) * T]; }
            v8f acc = {};
            acc = __builtin_amdgcn_wmma_f32_16x16x32_bf16(false, aq0, false, b0,
                                                          (short)0, acc, false, false);
            acc = __builtin_amdgcn_wmma_f32_16x16x32_bf16(false, aq1, false, b1,
                                                          (short)0, acc, false, false);
#pragma unroll
            for (int r = 0; r < 8; ++r)
                s[(r + hi * 8) * 1024 + kt * 16 + row] += acc[r] * 8.0f;
        }
        __syncthreads();

        // 3) row softmax in LDS: 16 lanes per row (group stays inside a wave)
        {
            const int r  = tid >> 4;             // 0..15
            const int c0 = tid & 15;             // 0..15
            float* sr = s + r * 1024;
            float m = -__builtin_inff();
            for (int c = c0; c < 1024; c += 16) m = fmaxf(m, sr[c]);
#pragma unroll
            for (int off = 8; off > 0; off >>= 1) m = fmaxf(m, __shfl_xor(m, off, 32));
            float sum = 0.f;
            for (int c = c0; c < 1024; c += 16) {
                float e = expf(sr[c] - m);
                sr[c] = e;
                sum += e;
            }
#pragma unroll
            for (int off = 8; off > 0; off >>= 1) sum += __shfl_xor(sum, off, 32);
            float inv = 1.0f / sum;
            for (int c = c0; c < 1024; c += 16) sr[c] *= inv;
        }
        __syncthreads();

        // 4) normalized attn tile -> global (coalesced float4), once
        {
            float4* dst4 = (float4*)(attn + ((((size_t)b * H + h) * T + q0) * T));
            const float4* s4 = (const float4*)s;
            for (int i = tid; i < 4096; i += 256) dst4[i] = s4[i];
        }

        // 5) PV from LDS: wave -> (d-slice = wave&3, K-half = wave>>2)
        {
            const int n0 = (wave & 3) * 16;
            const int kb = (wave >> 2) * 512;
            v8f acc = {};
            for (int k0 = kb; k0 < kb + 512; k0 += 32) {
                v16bf a, bf;
                const float* ap = s + row * 1024 + k0 + hi * 8;
#pragma unroll
                for (int j = 0; j < 8; ++j) { a[j] = (bf16_t)ap[j]; a[j + 8] = (bf16_t)ap[16 + j]; }
                const bf16_t* bp = Vp + (size_t)(k0 + hi * 8) * HD + n0 + row;
#pragma unroll
                for (int j = 0; j < 8; ++j) { bf[j] = bp[j * HD]; bf[j + 8] = bp[(16 + j) * HD]; }
                acc = __builtin_amdgcn_wmma_f32_16x16x32_bf16(false, a, false, bf,
                                                              (short)0, acc, false, false);
            }
#pragma unroll
            for (int r = 0; r < 8; ++r)
                pa[(wave * 16 + r + hi * 8) * 16 + row] = acc[r];
        }
        __syncthreads();
        if (wave < 4) {                      // combine K-halves, store ctx
            const int d = (wave & 3) * 16 + row;
#pragma unroll
            for (int r = 0; r < 8; ++r) {
                int m = r + hi * 8;
                float v = pa[(wave * 16 + m) * 16 + row] +
                          pa[((wave + 4) * 16 + m) * 16 + row];
                ctxb[((size_t)b * T + q0 + m) * E + h * HD + d] = (bf16_t)v;
            }
        }
        __syncthreads();                     // protect s/pa before next head
    }
}

// ---------------------------------------------------------------------------
// Kernel 3: out = ctx @ woT + bo   (f32 output)
// ---------------------------------------------------------------------------
__global__ void out_kernel(const bf16_t* __restrict__ A, const bf16_t* __restrict__ BT,
                           const float* __restrict__ bo, float* __restrict__ out) {
    const int lane = threadIdx.x & 31;
    const int wave = threadIdx.x >> 5;
    const int m0 = blockIdx.x * 16;
    const int n0 = (blockIdx.y * 8 + wave) * 16;
    const int row = lane & 15;
    const int hi  = lane >> 4;

    v8f acc = {};
    for (int k0 = 0; k0 < E; k0 += 32) {
        v16bf a, b;
        const bf16_t* ap = A + (size_t)(m0 + row) * E + k0 + hi * 8;
#pragma unroll
        for (int j = 0; j < 8; ++j) { a[j] = ap[j]; a[j + 8] = ap[16 + j]; }
        const bf16_t* bp = BT + (size_t)(k0 + hi * 8) * E + n0 + row;
#pragma unroll
        for (int j = 0; j < 8; ++j) { b[j] = bp[(size_t)j * E]; b[j + 8] = bp[(size_t)(16 + j) * E]; }
        acc = __builtin_amdgcn_wmma_f32_16x16x32_bf16(false, a, false, b,
                                                      (short)0, acc, false, false);
    }

    const int col = n0 + (lane & 15);
    const float bias = bo[col];
#pragma unroll
    for (int r = 0; r < 8; ++r) {
        int m = m0 + r + hi * 8;
        out[(size_t)m * E + col] = acc[r] + bias;
    }
}

// ---------------------------------------------------------------------------
// Host-side launcher
// ---------------------------------------------------------------------------
extern "C" void kernel_launch(void* const* d_in, const int* in_sizes, int n_in,
                              void* d_out, int out_size, void* d_ws, size_t ws_size,
                              hipStream_t stream) {
    const float*         x     = (const float*)d_in[0];
    const float*         u     = (const float*)d_in[1];
    const unsigned char* umask = (const unsigned char*)d_in[2];
    const float*         wq    = (const float*)d_in[3];
    const float*         wk    = (const float*)d_in[4];
    const float*         wv    = (const float*)d_in[5];
    const float*         wo    = (const float*)d_in[6];
    const float*         bo    = (const float*)d_in[7];
    const float*         wu    = (const float*)d_in[8];
    const float*         bu    = (const float*)d_in[9];

    // workspace layout (bytes)
    char* ws = (char*)d_ws;
    bf16_t* xb   = (bf16_t*)(ws + 0);            // 8192*512*2      =  8388608
    bf16_t* wqT  = (bf16_t*)(ws + 8388608);      // 512*512*2       =   524288
    bf16_t* wkT  = (bf16_t*)(ws + 8912896);
    bf16_t* wvT  = (bf16_t*)(ws + 9437184);
    bf16_t* woT  = (bf16_t*)(ws + 9961472);
    bf16_t* Qb   = (bf16_t*)(ws + 10485760);     // [b,h,t,d] bf16  =  8388608
    bf16_t* Ktb  = (bf16_t*)(ws + 18874368);     // [b,h,d,t] bf16
    bf16_t* Vb   = (bf16_t*)(ws + 27262976);     // [b,h,t,d] bf16
    bf16_t* ctxb = (bf16_t*)(ws + 35651584);     // [b,t,e]   bf16
    // total ws use: 44040192 bytes

    float* out  = (float*)d_out;                 // [8,1024,512]
    float* attn = out + (size_t)BS * T * E;      // [8,8,1024,1024]

    // 0) precision conversion / weight transpose
    cvt_x_kernel<<<(BS * T * E) / 256, 256, 0, stream>>>(x, xb);
    dim3 wt(32, 32);
    cvt_wT_kernel<<<wt, 256, 0, stream>>>(wq, wqT);
    cvt_wT_kernel<<<wt, 256, 0, stream>>>(wk, wkT);
    cvt_wT_kernel<<<wt, 256, 0, stream>>>(wv, wvT);
    cvt_wT_kernel<<<wt, 256, 0, stream>>>(wo, woT);

    // 1) Q, K, V projections (K stored transposed)
    dim3 pg(BS * T / 16, E / (16 * 8));
    proj_kernel<<<pg, 256, 0, stream>>>(xb, wqT, Qb, 0);
    proj_kernel<<<pg, 256, 0, stream>>>(xb, wkT, Ktb, 1);
    proj_kernel<<<pg, 256, 0, stream>>>(xb, wvT, Vb, 0);

    // 2) fused scores + softmax + PV (raw scores never leave LDS)
    dim3 ag(T / 16, BS);
    attn_fused_kernel<<<ag, 256, SM_TOT, stream>>>(Qb, Ktb, Vb, u, umask, wu, bu,
                                                   attn, ctxb);

    // 3) out = ctx @ woT + bo
    out_kernel<<<pg, 256, 0, stream>>>(ctxb, woT, bo, out);
}
